// Decoder_61873298866480
// MI455X (gfx1250) — matrix operand order repaired and checked by reading, not compiled, not run on hardware
//
#include <hip/hip_runtime.h>
#include <hip/hip_bf16.h>
#include <math.h>

typedef __attribute__((ext_vector_type(16))) __bf16 v16bf;
typedef __attribute__((ext_vector_type(8)))  float  v8f;
typedef __attribute__((ext_vector_type(2)))  float  v2f;

#define T_STEPS 512
#define B_DIM   128
#define D_DIM   512
#define V_DIM   33
#define E_DIM   128
#define G4D     2048   // 4*D

// ---------------- init: g0 = b_emb @ W_ih + b_ih + b_hh ----------------
__global__ void k_g0(const float* __restrict__ b_emb,
                     const float* __restrict__ W_ih,
                     const float* __restrict__ b_ih,
                     const float* __restrict__ b_hh,
                     float* __restrict__ g0) {
  int n = blockIdx.x * blockDim.x + threadIdx.x;
  if (n >= G4D) return;
  float s = b_ih[n] + b_hh[n];
  for (int e = 0; e < E_DIM; ++e) s += b_emb[e] * W_ih[e * G4D + n];
  g0[n] = s;
}

// ---------------- init: swizzle W_hh -> bf16 hi/lo B-fragments ----------------
// B-frag (16x16x32 bf16 B operand, 32x16 tile): idx = ((kb*128 + nb)*32 + lane)*16 + e
// lane holds column nb*16 + (lane&15); element e -> K = kb*32 + e + (lane>=16 ? 16 : 0)
__global__ void k_wswz(const float* __restrict__ W_hh,
                       __bf16* __restrict__ Whi, __bf16* __restrict__ Wlo) {
  int idx = blockIdx.x * blockDim.x + threadIdx.x;
  if (idx >= D_DIM * G4D) return;
  int e  = idx & 15;
  int l  = (idx >> 4) & 31;
  int nb = (idx >> 9) & 127;
  int kb = idx >> 16;
  int K  = kb * 32 + e + ((l & 16) ? 16 : 0);
  int N  = nb * 16 + (l & 15);
  float w = W_hh[K * G4D + N];
  __bf16 hi = (__bf16)w;
  Whi[idx] = hi;
  Wlo[idx] = (__bf16)(w - (float)hi);
}

// ---------------- init: zero state buffers, transpose W_out ----------------
__global__ void k_init(float* h, float* c, float* z, float* WoutT,
                       const float* __restrict__ W_out,
                       __bf16* F0hi, __bf16* F0lo) {
  const int total = 65536 + 65536 + 4224 + 33792 + 65536 + 65536;
  for (int i = blockIdx.x * blockDim.x + threadIdx.x; i < total;
       i += gridDim.x * blockDim.x) {
    int j = i;
    if (j < 65536) { h[j] = 0.f; continue; } j -= 65536;
    if (j < 65536) { c[j] = 0.f; continue; } j -= 65536;
    if (j < 4224)  { z[j] = 0.f; continue; } j -= 4224;
    if (j < 33792) { int v = j >> 10, k = j & 1023; WoutT[v * 1024 + k] = W_out[k * V_DIM + v]; continue; } j -= 33792;
    if (j < 65536) { F0hi[j] = (__bf16)0.f; continue; } j -= 65536;
    F0lo[j] = (__bf16)0.f;
  }
}

// ---------------- per-step K1: gates GEMM (split-bf16 WMMA) + LSTM cell ----------------
// A-frag (16x32 bf16 A operand): idx = ((mb*16 + kb)*32 + lane)*16 + e
// lane&15 = row M; element e -> K = e + (lane>=16 ? 16 : 0)
__global__ __launch_bounds__(256) void k_gates(
    const __bf16* __restrict__ Ahi_r, const __bf16* __restrict__ Alo_r,
    __bf16* __restrict__ Ahi_w, __bf16* __restrict__ Alo_w,
    const __bf16* __restrict__ Whi, const __bf16* __restrict__ Wlo,
    const float* __restrict__ g0,
    float* __restrict__ c, float* __restrict__ h,
    float* __restrict__ sumexp, float* __restrict__ sumraw) {
  const int tid   = threadIdx.x;
  const int wave  = tid >> 5;      // 0..7 -> row block mb
  const int lane  = tid & 31;
  const int chunk = blockIdx.x;    // 0..31 -> gate-local columns j0..j0+15
  const int j0    = chunk * 16;

  if (blockIdx.x == 0) {   // reset attention accumulators for this step
    for (int i = tid; i < D_DIM; i += 256) { sumexp[i] = 0.f; sumraw[i] = 0.f; }
  }

  v8f acc[4] = {};
  for (int kb = 0; kb < 16; ++kb) {
    const v16bf ahi = *(const v16bf*)(Ahi_r + (((wave * 16 + kb) * 32 + lane) << 4));
    const v16bf alo = *(const v16bf*)(Alo_r + (((wave * 16 + kb) * 32 + lane) << 4));
#pragma unroll
    for (int gb = 0; gb < 4; ++gb) {
      const int nbG = gb * 32 + chunk;
      const v16bf bhi = *(const v16bf*)(Whi + (((kb * 128 + nbG) * 32 + lane) << 4));
      const v16bf blo = *(const v16bf*)(Wlo + (((kb * 128 + nbG) * 32 + lane) << 4));
      acc[gb] = __builtin_amdgcn_wmma_f32_16x16x32_bf16(false, ahi, false, bhi, (short)0, acc[gb], false, false);
      acc[gb] = __builtin_amdgcn_wmma_f32_16x16x32_bf16(false, ahi, false, blo, (short)0, acc[gb], false, false);
      acc[gb] = __builtin_amdgcn_wmma_f32_16x16x32_bf16(false, alo, false, bhi, (short)0, acc[gb], false, false);
    }
  }

  const int n      = lane & 15;
  const int hihalf = lane >> 4;
  const int j      = j0 + n;
  const float bi = g0[0 * D_DIM + j];
  const float bf = g0[1 * D_DIM + j];
  const float bg = g0[2 * D_DIM + j];
  const float bo = g0[3 * D_DIM + j];

  // A-fragment coordinates for next step's h (d = j), uniform per lane
  const int k32 = j & 31;
  const int e  = k32 & 15;
  const int lh = k32 >> 4;
  const int kb = j >> 5;

#pragma unroll
  for (int r = 0; r < 8; ++r) {
    const int b = wave * 16 + r + hihalf * 8;
    float I = acc[0][r] + bi;
    float F = acc[1][r] + bf;
    float G = acc[2][r] + bg;
    float O = acc[3][r] + bo;
    I = 1.f / (1.f + expf(-I));
    F = 1.f / (1.f + expf(-F));
    O = 1.f / (1.f + expf(-O));
    const float cp = c[b * D_DIM + j];
    const float cn = F * cp + I * tanhf(G);
    const float hn = O * tanhf(cn);
    c[b * D_DIM + j] = cn;
    h[b * D_DIM + j] = hn;
    const int la  = (b & 15) + lh * 16;
    const int mb  = b >> 4;
    const int off = ((mb * 16 + kb) * 32 + la) * 16 + e;
    const __bf16 hh = (__bf16)hn;
    Ahi_w[off] = hh;
    Alo_w[off] = (__bf16)(hn - (float)hh);
  }
}

// ---------------- per-step K2: score = hidden^T @ h (f32 WMMA), col sums ----------------
__global__ __launch_bounds__(128) void k_score(
    const float* __restrict__ hidden,   // enc_h + t*B*D
    const float* __restrict__ h,
    float* __restrict__ sumexp, float* __restrict__ sumraw) {
  __shared__ float Ah[64 * 68];   // Ah[m][k] = hidden[b=k][r0+m], pitch 68 (bank-safe)
  __shared__ float Bh[64 * 68];   // Bh[n][k] = h[b=k][c0+n]
  __shared__ float colE[64];
  __shared__ float colR[64];
  const int tid  = threadIdx.x;
  const int wave = tid >> 5;      // 4 waves, each 16 rows x 64 cols strip
  const int lane = tid & 31;
  const int rr = blockIdx.x >> 3, cc = blockIdx.x & 7;
  const int r0 = rr * 64, c0 = cc * 64;
  if (tid < 64) { colE[tid] = 0.f; colR[tid] = 0.f; }

  v8f acc[4] = {};
  const int mloc = wave * 16 + (lane & 15);
  const int koff = (lane >> 4) * 2;   // f32 A/B: lanes 16-31 hold K+2,K+3

  for (int kc = 0; kc < 2; ++kc) {    // K=128 staged in halves of 64
    __syncthreads();
    for (int i = tid; i < 64 * 64; i += 128) {
      const int m = i & 63, bl = i >> 6, b = kc * 64 + bl;
      Ah[m * 68 + bl] = hidden[b * D_DIM + r0 + m];
      Bh[m * 68 + bl] = h[b * D_DIM + c0 + m];
    }
    __syncthreads();
    for (int kk = 0; kk < 16; ++kk) {
      const int k = kk * 4 + koff;
      const v2f a = *(const v2f*)(Ah + mloc * 68 + k);
#pragma unroll
      for (int tt = 0; tt < 4; ++tt) {
        const int nl = tt * 16 + (lane & 15);
        const v2f bv = *(const v2f*)(Bh + nl * 68 + k);
        acc[tt] = __builtin_amdgcn_wmma_f32_16x16x4_f32(false, a, false, bv, (short)0, acc[tt], false, false);
      }
    }
  }

#pragma unroll
  for (int tt = 0; tt < 4; ++tt) {
    float se = 0.f, sr = 0.f;
#pragma unroll
    for (int r = 0; r < 8; ++r) { const float v = acc[tt][r]; se += expf(v); sr += v; }
    se += __shfl_xor(se, 16, 32);   // fold rows 8..15 half onto 0..7 half
    sr += __shfl_xor(sr, 16, 32);
    if (lane < 16) {
      atomicAdd(&colE[tt * 16 + lane], se);
      atomicAdd(&colR[tt * 16 + lane], sr);
    }
  }
  __syncthreads();
  if (tid < 64) {
    atomicAdd(&sumexp[c0 + tid], colE[tid]);
    atomicAdd(&sumraw[c0 + tid], colR[tid]);
  }
}

// ---------------- per-step K3: w, incremental z, logits, log_softmax ----------------
__global__ __launch_bounds__(128) void k_out(
    const float* __restrict__ hidden,
    const float* __restrict__ h,
    const float* __restrict__ sumexp, const float* __restrict__ sumraw,
    const float* __restrict__ WoutT, const float* __restrict__ b_out,
    float* __restrict__ z, float* __restrict__ out_t) {
  __shared__ float hrow[D_DIM];
  __shared__ float whid[D_DIM];
  __shared__ float lg[40];
  const int b = blockIdx.x;
  const int tid = threadIdx.x;
  for (int j = tid; j < D_DIM; j += 128) {
    const float w = sumexp[j] / sumraw[j];
    hrow[j] = h[b * D_DIM + j];
    whid[j] = w * hidden[b * D_DIM + j];
  }
  __syncthreads();
  if (tid < V_DIM) {
    const float* wt = WoutT + tid * 1024;
    float dh = 0.f, dc = 0.f;
    for (int j = 0; j < D_DIM; ++j) dh += hrow[j] * wt[j];
    for (int j = 0; j < D_DIM; ++j) dc += whid[j] * wt[512 + j];
    const float zn = z[b * V_DIM + tid] + dc;   // z = c_t @ W_out_bot, maintained incrementally
    z[b * V_DIM + tid] = zn;
    lg[tid] = dh + zn + b_out[tid];
  }
  __syncthreads();
  if (tid < V_DIM) {
    float m = -1e30f;
    for (int v = 0; v < V_DIM; ++v) m = fmaxf(m, lg[v]);
    float s = 0.f;
    for (int v = 0; v < V_DIM; ++v) s += expf(lg[v] - m);
    out_t[b * V_DIM + tid] = lg[tid] - m - logf(s);
  }
}

extern "C" void kernel_launch(void* const* d_in, const int* in_sizes, int n_in,
                              void* d_out, int out_size, void* d_ws, size_t ws_size,
                              hipStream_t stream) {
  (void)in_sizes; (void)n_in; (void)out_size; (void)ws_size;
  const float* enc_h = (const float*)d_in[0];
  const float* b_emb = (const float*)d_in[2];
  const float* W_ih  = (const float*)d_in[3];
  const float* W_hh  = (const float*)d_in[4];
  const float* b_ih  = (const float*)d_in[5];
  const float* b_hh  = (const float*)d_in[6];
  const float* W_out = (const float*)d_in[7];
  const float* b_out = (const float*)d_in[8];
  float* out = (float*)d_out;

  char* p = (char*)d_ws;
  auto carve = [&](size_t bytes) { char* r = p; p += (bytes + 255) & ~(size_t)255; return r; };
  __bf16* Whi   = (__bf16*)carve(sizeof(__bf16) * (size_t)D_DIM * G4D);   // 2 MB
  __bf16* Wlo   = (__bf16*)carve(sizeof(__bf16) * (size_t)D_DIM * G4D);   // 2 MB
  __bf16* Fhi0  = (__bf16*)carve(2 * 65536);
  __bf16* Fhi1  = (__bf16*)carve(2 * 65536);
  __bf16* Flo0  = (__bf16*)carve(2 * 65536);
  __bf16* Flo1  = (__bf16*)carve(2 * 65536);
  float*  h     = (float*)carve(4 * 65536);
  float*  c     = (float*)carve(4 * 65536);
  float*  g0    = (float*)carve(4 * 2048);
  float*  z     = (float*)carve(4 * 4224);
  float*  WoutT = (float*)carve(4 * 33792);
  float*  sumexp = (float*)carve(4 * 512);
  float*  sumraw = (float*)carve(4 * 512);
  __bf16* Fhi[2] = {Fhi0, Fhi1};
  __bf16* Flo[2] = {Flo0, Flo1};

  k_g0  <<<8,    256, 0, stream>>>(b_emb, W_ih, b_ih, b_hh, g0);
  k_wswz<<<4096, 256, 0, stream>>>(W_hh, Whi, Wlo);
  k_init<<<300,  256, 0, stream>>>(h, c, z, WoutT, W_out, Fhi[0], Flo[0]);

  for (int t = 0; t < T_STEPS; ++t) {
    const float* hid = enc_h + (size_t)t * B_DIM * D_DIM;
    k_gates<<<32, 256, 0, stream>>>(Fhi[t & 1], Flo[t & 1], Fhi[(t + 1) & 1], Flo[(t + 1) & 1],
                                    Whi, Wlo, g0, c, h, sumexp, sumraw);
    k_score<<<64, 128, 0, stream>>>(hid, h, sumexp, sumraw);
    k_out  <<<128, 128, 0, stream>>>(hid, h, sumexp, sumraw, WoutT, b_out, z,
                                     out + (size_t)t * B_DIM * V_DIM);
  }
}
